// MHA_17016660426968
// MI455X (gfx1250) — compile-verified, matrix operand order checked by reading, not run on hardware
//
#include <hip/hip_runtime.h>
#include <hip/hip_bf16.h>

typedef __attribute__((ext_vector_type(16))) __bf16 v16bf;
typedef __attribute__((ext_vector_type(8)))  float  v8f;
typedef __bf16 bf16;

#define BB 2
#define SS 1024
#define DD 4096
#define HH 32
#define HD 128

#define WMMA_BF16(a, b, c) \
  __builtin_amdgcn_wmma_f32_16x16x32_bf16(false, (a), false, (b), (short)0, (c), false, false)

// async global -> LDS copy of 16 bytes (per lane); tracked by ASYNCcnt
#define ASYNC_LDS_B128(ldsoff_u32, gaddr_u64)                                  \
  asm volatile("global_load_async_to_lds_b128 %0, %1, off" ::"v"(ldsoff_u32),  \
               "v"(gaddr_u64)                                                  \
               : "memory")
#define WAIT_ASYNC0() asm volatile("s_wait_asynccnt 0x0" ::: "memory")
#define WAIT_DS0()    asm volatile("s_wait_dscnt 0x0" ::: "memory")

union FragA {
  v16bf v;
  uint4 q[2];
  bf16  h[16];
};

__device__ __forceinline__ unsigned lds_off(const void* p) {
  return (unsigned)(size_t)p;  // low 32 bits of flat LDS address = LDS byte offset
}

// ---------------------------------------------------------------------------
// GEMM: Out[M,N] = A[M,K] * W[N,K]^T   (bf16 WMMA, f32 accumulate)
// Block tile 128x128, BK=32, 8 waves (2x4), each wave 64x32 (4x2 WMMA frags).
// ---------------------------------------------------------------------------
template <bool A_BF16, bool OUT_F32>
__global__ __launch_bounds__(256) void gemm_wmma_nt(const void* __restrict__ Aptr,
                                                    const float* __restrict__ W,
                                                    void* __restrict__ Out,
                                                    int M, int N, int K) {
  constexpr int BM = 128, BN = 128, BK = 32;
  constexpr int SA = 40, SB = 40;  // LDS row strides (elements); 80B rows, 16B aligned
  __shared__ bf16 As[BM * SA];
  __shared__ bf16 Bs[BN * SB];

  const int t    = threadIdx.x;
  const int lane = t & 31;
  const int w    = t >> 5;
  const int wm   = w & 1;   // 2 wave rows (64 each)
  const int wn   = w >> 1;  // 4 wave cols (32 each)
  const int lh   = lane & 15;
  const int kb   = (lane >> 4) ? 8 : 0;  // K-chunk base for A/B fragments
  const int hb   = (lane >> 4) ? 8 : 0;  // C/D row half

  const int mBase = blockIdx.y * BM;
  const int nBase = blockIdx.x * BN;

  // staging assignments: each thread owns one 16-element half-row of A and of B
  const int srow = t >> 1;          // 0..127
  const int scol = (t & 1) * 16;    // 0 or 16

  const v8f vzero = {0.f, 0.f, 0.f, 0.f, 0.f, 0.f, 0.f, 0.f};
  v8f acc[4][2];
#pragma unroll
  for (int i = 0; i < 4; i++)
#pragma unroll
    for (int j = 0; j < 2; j++) acc[i][j] = vzero;

  for (int ko = 0; ko < K; ko += BK) {
    __syncthreads();
    // ---- stage A tile (BM x BK) as bf16 ----
    if constexpr (A_BF16) {
      const bf16* src = (const bf16*)Aptr + (size_t)(mBase + srow) * K + ko + scol;
      const unsigned d0 = lds_off(&As[srow * SA + scol]);
      ASYNC_LDS_B128(d0, (unsigned long long)(size_t)src);
      ASYNC_LDS_B128(d0 + 16u, (unsigned long long)(size_t)(src + 8));
    } else {
      const float* src = (const float*)Aptr + (size_t)(mBase + srow) * K + ko + scol;
      if (ko + BK < K) __builtin_prefetch(src + BK, 0, 1);
#pragma unroll
      for (int c = 0; c < 16; c += 4) {
        float4 f = *(const float4*)(src + c);
        bf16* dst = &As[srow * SA + scol + c];
        dst[0] = (bf16)f.x; dst[1] = (bf16)f.y; dst[2] = (bf16)f.z; dst[3] = (bf16)f.w;
      }
    }
    // ---- stage B tile (BN x BK) from W rows (convert f32 -> bf16) ----
    {
      const float* src = W + (size_t)(nBase + srow) * K + ko + scol;
      if (ko + BK < K) __builtin_prefetch(src + BK, 0, 1);
#pragma unroll
      for (int c = 0; c < 16; c += 4) {
        float4 f = *(const float4*)(src + c);
        bf16* dst = &Bs[srow * SB + scol + c];
        dst[0] = (bf16)f.x; dst[1] = (bf16)f.y; dst[2] = (bf16)f.z; dst[3] = (bf16)f.w;
      }
    }
    if constexpr (A_BF16) WAIT_ASYNC0();
    __syncthreads();

    // ---- fragments + WMMA ----
    FragA a[4], b[2];
#pragma unroll
    for (int i = 0; i < 4; i++) {
      const bf16* ap = &As[(wm * 64 + i * 16 + lh) * SA + kb];
      a[i].q[0] = *(const uint4*)ap;
      a[i].q[1] = *(const uint4*)(ap + 16);
    }
#pragma unroll
    for (int j = 0; j < 2; j++) {
      const bf16* bp = &Bs[(wn * 32 + j * 16 + lh) * SB + kb];
      b[j].q[0] = *(const uint4*)bp;
      b[j].q[1] = *(const uint4*)(bp + 16);
    }
#pragma unroll
    for (int i = 0; i < 4; i++)
#pragma unroll
      for (int j = 0; j < 2; j++) acc[i][j] = WMMA_BF16(a[i].v, b[j].v, acc[i][j]);
  }

  // ---- epilogue: C/D layout M = v + 8*(lane>=16), N = lane&15 ----
#pragma unroll
  for (int i = 0; i < 4; i++)
#pragma unroll
    for (int j = 0; j < 2; j++)
#pragma unroll
      for (int v = 0; v < 8; v++) {
        const int m = mBase + wm * 64 + i * 16 + hb + v;
        const int n = nBase + wn * 32 + j * 16 + lh;
        const float val = acc[i][j][v];
        if constexpr (OUT_F32)
          ((float*)Out)[(size_t)m * N + n] = val;
        else
          ((bf16*)Out)[(size_t)m * N + n] = (bf16)val;
      }
}

// ---------------------------------------------------------------------------
// RoPE (in place on bf16 Q and K), uses the harness-provided cos/sin tables.
// ---------------------------------------------------------------------------
__global__ void rope_kernel(bf16* __restrict__ Q, bf16* __restrict__ Kc,
                            const float* __restrict__ fcos, const float* __restrict__ fsin) {
  const int PH = HD / 2;  // 64
  int idx = blockIdx.x * blockDim.x + threadIdx.x;
  if (idx >= BB * SS * HH * PH) return;
  int p = idx % PH;
  int h = (idx / PH) % HH;
  int s = (idx / (PH * HH)) % SS;
  int b = idx / (PH * HH * SS);
  float c  = fcos[s * PH + p];
  float sn = fsin[s * PH + p];
  size_t base = (((size_t)b * SS + s) * HH + h) * HD + 2 * p;
  float qe = (float)Q[base], qo = (float)Q[base + 1];
  Q[base]     = (bf16)(qe * c - qo * sn);
  Q[base + 1] = (bf16)(qe * sn + qo * c);
  float ke = (float)Kc[base], ko = (float)Kc[base + 1];
  Kc[base]     = (bf16)(ke * c - ko * sn);
  Kc[base + 1] = (bf16)(ke * sn + ko * c);
}

// ---------------------------------------------------------------------------
// Flash-style attention. Grid (S/128, H, B), 256 threads.
// Each wave owns 16 query rows; key blocks of 32; online softmax.
// K block staged to LDS via async-to-LDS (shared by all 8 waves);
// V block staged transposed for contiguous P*V B-fragments.
// ---------------------------------------------------------------------------
__global__ __launch_bounds__(256) void attn_kernel(const bf16* __restrict__ Q,
                                                   const bf16* __restrict__ Kc,
                                                   const bf16* __restrict__ Vc,
                                                   bf16* __restrict__ Ctx) {
  constexpr int SK = 136;  // K tile row stride (elements); 272B = 17x16B
  constexpr int SV = 40;   // Vt row stride
  constexpr int SP = 40;   // P row stride
  __shared__ bf16 Ks[32 * SK];      // K block, row-major [key][hd]
  __shared__ bf16 Vt[HD * SV];      // transposed V block [hd][key]
  __shared__ bf16 Pl[8 * 16 * SP];  // per-wave 16x32 probability tile

  const int t = threadIdx.x, lane = t & 31, w = t >> 5;
  const int lh = lane & 15;
  const int hb = (lane >> 4) ? 8 : 0;
  const int b = blockIdx.z, h = blockIdx.y;
  const int q0 = blockIdx.x * 128 + w * 16;
  const float scale = 0.08838834764831845f;  // 1/sqrt(128)

  const size_t rowstride = (size_t)HH * HD;  // 4096
  const size_t bh_off = ((size_t)b * SS) * rowstride + (size_t)h * HD;

  // preload Q A-fragments (16 rows x 128 hd = 4 K-steps)
  FragA qa[4];
  {
    const bf16* qrow = Q + bh_off + (size_t)(q0 + lh) * rowstride;
#pragma unroll
    for (int ks = 0; ks < 4; ks++) {
      const int kk = ks * 32 + hb;
      qa[ks].q[0] = *(const uint4*)(qrow + kk);
      qa[ks].q[1] = *(const uint4*)(qrow + kk + 16);
    }
  }

  const v8f vzero = {0.f, 0.f, 0.f, 0.f, 0.f, 0.f, 0.f, 0.f};
  v8f cacc[8];
#pragma unroll
  for (int n = 0; n < 8; n++) cacc[n] = vzero;
  float mrun[8], lrun[8];
#pragma unroll
  for (int v = 0; v < 8; v++) { mrun[v] = -1e30f; lrun[v] = 0.f; }

  const int nkb = blockIdx.x * 4 + 4;  // causal: keys up to end of this q tile
  const int vkey = t >> 3;             // 0..31
  const int vhd  = (t & 7) * 16;       // 0..112

  for (int kb_i = 0; kb_i < nkb; kb_i++) {
    const int kbase = kb_i * 32;
    __syncthreads();
    // ---- async-stage K block (32 keys x 128 hd), row-major, 2x16B per thread
#pragma unroll
    for (int c = 0; c < 2; c++) {
      const int cid  = t * 2 + c;
      const int krow = cid >> 4;         // 0..31
      const int kcol = (cid & 15) * 8;   // 0..120 (elements)
      const bf16* gsrc = Kc + bh_off + (size_t)(kbase + krow) * rowstride + kcol;
      ASYNC_LDS_B128(lds_off(&Ks[krow * SK + kcol]),
                     (unsigned long long)(size_t)gsrc);
    }
    // ---- stage V block transposed into LDS: Vt[hd][key]
    {
      const bf16* vrow = Vc + bh_off + (size_t)(kbase + vkey) * rowstride + vhd;
#pragma unroll
      for (int e = 0; e < 16; e++) Vt[(vhd + e) * SV + vkey] = vrow[e];
    }
    WAIT_ASYNC0();
    __syncthreads();

    // ---- S = Q * K^T (16 x 32) ----
    v8f sacc[2];
    sacc[0] = vzero; sacc[1] = vzero;
#pragma unroll
    for (int j = 0; j < 2; j++) {
      const bf16* krow = &Ks[(j * 16 + lh) * SK];
#pragma unroll
      for (int ks = 0; ks < 4; ks++) {
        FragA kf;
        const int kk = ks * 32 + hb;
        kf.q[0] = *(const uint4*)(krow + kk);
        kf.q[1] = *(const uint4*)(krow + kk + 16);
        sacc[j] = WMMA_BF16(qa[ks].v, kf.v, sacc[j]);
      }
    }

    // ---- mask + online softmax (row reductions across 16-lane halves) ----
    float p[2][8];
#pragma unroll
    for (int v = 0; v < 8; v++) {
      const int qrow = q0 + hb + v;
      float x0 = sacc[0][v] * scale + ((kbase + lh)      > qrow ? -1e9f : 0.f);
      float x1 = sacc[1][v] * scale + ((kbase + 16 + lh) > qrow ? -1e9f : 0.f);
      float m = fmaxf(x0, x1);
      for (int d = 1; d < 16; d <<= 1) m = fmaxf(m, __shfl_xor(m, d, 32));
      const float mnew = fmaxf(mrun[v], m);
      const float f  = __expf(mrun[v] - mnew);
      const float p0 = __expf(x0 - mnew);
      const float p1 = __expf(x1 - mnew);
      float rs = p0 + p1;
      for (int d = 1; d < 16; d <<= 1) rs += __shfl_xor(rs, d, 32);
      lrun[v] = lrun[v] * f + rs;
      mrun[v] = mnew;
      p[0][v] = p0; p[1][v] = p1;
#pragma unroll
      for (int n = 0; n < 8; n++) cacc[n][v] *= f;
    }

    // ---- transpose P through LDS: D-layout -> A-layout ----
    bf16* pw = &Pl[w * 16 * SP];
#pragma unroll
    for (int j = 0; j < 2; j++)
#pragma unroll
      for (int v = 0; v < 8; v++) pw[(hb + v) * SP + j * 16 + lh] = (bf16)p[j][v];
    WAIT_DS0();  // intra-wave LDS RAW fence (CDNA5 split DS counter)

    FragA pa;
    {
      const bf16* pp = &pw[lh * SP + hb];
      pa.q[0] = *(const uint4*)pp;
      pa.q[1] = *(const uint4*)(pp + 16);
    }
    // ---- Ctx += P * V ----
#pragma unroll
    for (int n = 0; n < 8; n++) {
      FragA vf;
      const bf16* vp = &Vt[(n * 16 + lh) * SV + hb];
      vf.q[0] = *(const uint4*)vp;
      vf.q[1] = *(const uint4*)(vp + 16);
      cacc[n] = WMMA_BF16(pa.v, vf.v, cacc[n]);
    }
  }

  // ---- normalize and write context (bf16, [b,s,h,hd]) ----
#pragma unroll
  for (int v = 0; v < 8; v++) {
    const float linv = 1.0f / lrun[v];
    const int qrow = q0 + hb + v;
    bf16* crow = Ctx + bh_off + (size_t)qrow * rowstride;
#pragma unroll
    for (int n = 0; n < 8; n++) crow[n * 16 + lh] = (bf16)(cacc[n][v] * linv);
  }
}

// ---------------------------------------------------------------------------
extern "C" void kernel_launch(void* const* d_in, const int* in_sizes, int n_in,
                              void* d_out, int out_size, void* d_ws, size_t ws_size,
                              hipStream_t stream) {
  (void)in_sizes; (void)n_in; (void)out_size; (void)ws_size;
  const float* x    = (const float*)d_in[0];
  const float* fcos = (const float*)d_in[1];
  const float* fsin = (const float*)d_in[2];
  const float* wq   = (const float*)d_in[6];
  const float* wk   = (const float*)d_in[7];
  const float* wv   = (const float*)d_in[8];
  const float* wo   = (const float*)d_in[9];
  float* out = (float*)d_out;

  const size_t BUF = (size_t)BB * SS * DD * sizeof(bf16);  // 16 MiB
  char* ws = (char*)d_ws;
  bf16* qb = (bf16*)(ws + 0 * BUF);
  bf16* kb = (bf16*)(ws + 1 * BUF);
  bf16* vb = (bf16*)(ws + 2 * BUF);
  bf16* cb = (bf16*)(ws + 3 * BUF);

  const int M = BB * SS;          // 2048
  dim3 gg(DD / 128, M / 128);     // (32, 16)
  gemm_wmma_nt<false, false><<<gg, 256, 0, stream>>>(x, wq, qb, M, DD, DD);
  gemm_wmma_nt<false, false><<<gg, 256, 0, stream>>>(x, wk, kb, M, DD, DD);
  gemm_wmma_nt<false, false><<<gg, 256, 0, stream>>>(x, wv, vb, M, DD, DD);

  const int nrope = BB * SS * HH * (HD / 2);
  rope_kernel<<<(nrope + 255) / 256, 256, 0, stream>>>(qb, kb, fcos, fsin);

  attn_kernel<<<dim3(SS / 128, HH, BB), 256, 0, stream>>>(qb, kb, vb, cb);

  gemm_wmma_nt<true, true><<<gg, 256, 0, stream>>>(cb, wo, out, M, DD, DD);
}
